// LPN_36292473651320
// MI455X (gfx1250) — matrix operand order repaired and checked by reading, not compiled, not run on hardware
//
#include <hip/hip_runtime.h>
#include <hip/hip_bf16.h>

// ---------------------------------------------------------------------------
// Types for gfx1250 WMMA (wave32)
// ---------------------------------------------------------------------------
typedef __bf16 bf16;
typedef bf16  bf16x16 __attribute__((ext_vector_type(16)));
typedef float f32x8   __attribute__((ext_vector_type(8)));

#define CH_     192
#define GROUPS_ 32
#define ZSCALE_ 5.0f

// ---------------------------------------------------------------------------
// f32 -> bf16 cast
// ---------------------------------------------------------------------------
__global__ void lpn_cast_bf16(const float* __restrict__ in, bf16* __restrict__ out, int n) {
  int i = blockIdx.x * blockDim.x + threadIdx.x;
  if (i < n) out[i] = (bf16)in[i];
}

// f32 [K][N] -> bf16 [N][K] (weights HWIO -> N-major for direct B-frag loads)
__global__ void lpn_cast_transpose(const float* __restrict__ w, bf16* __restrict__ wT,
                                   int K, int N) {
  int i = blockIdx.x * blockDim.x + threadIdx.x;
  if (i >= K * N) return;
  int k = i / N, n = i - k * N;
  wT[(size_t)n * K + k] = (bf16)w[i];
}

__global__ void lpn_zero_f32(float* __restrict__ p, int n) {
  int i = blockIdx.x * blockDim.x + threadIdx.x;
  if (i < n) p[i] = 0.0f;
}

// ---------------------------------------------------------------------------
// Implicit-GEMM 3x3 SAME conv, bf16 in/wgt, f32 WMMA accum, bias add.
//   act:  [D,H,W,Cin] bf16
//   wgtT: [192][9*Cin] bf16 (transposed; B-frags loaded straight from L2/WGP$)
//   out:  [D,H,W,192] bf16
// GEMM: M = D*H*W (multiple of 128), N = 192, K = 9*Cin.
// Block: 256 thr = 8 waves; block tile 128(M) x 64(N); wave = 16x64 strip =
// 4 accumulators -> 4 v_wmma per K-chunk with A-frag reuse x4.
// A tile is staged with CDNA5 async copies (GLOBAL_LOAD_ASYNC_TO_LDS_B128,
// ASYNCcnt) into a quad-buffered LDS ring, issue-ahead = 2 chunks:
//   writer = chunk kc+2, slowest reader = chunk kc-1 (3 apart) => 4 buffers.
// ---------------------------------------------------------------------------
template <int CIN>
__global__ __launch_bounds__(256, 2)
void lpn_conv3x3_wmma(const bf16* __restrict__ act, const bf16* __restrict__ wgtT,
                      const float* __restrict__ bias, bf16* __restrict__ out,
                      int H, int W, int M)
{
  constexpr int K9   = 9 * CIN;
  constexpr int NK   = K9 / 32;        // K chunks of 32
  constexpr int AST  = 40;             // bf16 row stride (32 + pad, multiple of 8)
  constexpr unsigned ABYTES = 128u * AST * sizeof(bf16);   // 10240 B per buffer

  __shared__ __align__(16) bf16 Ab[4][128 * AST];

  const int tid   = threadIdx.x;
  const int nbase = blockIdx.y * 64;
  const int mbase = blockIdx.x * 128;

  // --- A staging mapping: 128 rows x 32 ks; thread = 16 contiguous ks (32 B)
  const int ar = tid & 127;            // row within tile
  const int ah = tid >> 7;             // 0/1 : k-half
  const int gm = mbase + ar;           // global output row (< M always)
  const int xx = gm % W;
  const int yy = (gm / W) % H;
  const int dd = gm / (W * H);

  const unsigned ldsbase = (unsigned)(size_t)&Ab[0][ar * AST + ah * 16];
  const uint4 z4 = {0u, 0u, 0u, 0u};

  // issue async copy of A chunk kc2 into LDS ring buffer (kc2 & 3)
  auto issueA = [&](int kc2) {
    const int tb    = kc2 & 3;
    const int kb    = kc2 << 5;
    const int tap   = kb / CIN;        // constexpr CIN: folds to shifts/muls
    const int cbase = kb - tap * CIN;
    const int ys = yy + tap / 3 - 1;
    const int xs = xx + tap % 3 - 1;
    const unsigned lo = ldsbase + (unsigned)tb * ABYTES;
    if ((unsigned)ys < (unsigned)H && (unsigned)xs < (unsigned)W) {
      const unsigned long long ga = (unsigned long long)(size_t)
          (act + (size_t)((dd * H + ys) * W + xs) * CIN + cbase + ah * 16);
      asm volatile("global_load_async_to_lds_b128 %0, %1, off"
                   :: "v"(lo), "v"(ga) : "memory");
      asm volatile("global_load_async_to_lds_b128 %0, %1, off"
                   :: "v"(lo + 16u), "v"(ga + 16ull) : "memory");
    } else {
      *(uint4*)&Ab[tb][ar * AST + ah * 16]     = z4;
      *(uint4*)&Ab[tb][ar * AST + ah * 16 + 8] = z4;
    }
  };

  // --- wave fragment mapping (ISA 7.12.2: 16-bit A 16x32 / B 32x16)
  const int lane = tid & 31;
  const int wv   = tid >> 5;           // 8 waves along M
  const int mrow = lane & 15;
  const int kh   = lane >> 4;          // 0/1

  f32x8 acc0 = {0.f,0.f,0.f,0.f,0.f,0.f,0.f,0.f};
  f32x8 acc1 = acc0, acc2 = acc0, acc3 = acc0;

  // ---- pipeline prologue: 2 chunks in flight
  issueA(0);
  issueA(1);

  for (int kc = 0; kc < NK; ++kc) {
    if (kc + 2 < NK) {
      issueA(kc + 2);
      // chunks kc+1, kc+2 (2 x b128 each) may stay outstanding
      asm volatile("s_wait_asynccnt 0x4" ::: "memory");
    } else {
      asm volatile("s_wait_asynccnt 0x0" ::: "memory");
    }
    __syncthreads();

    // A frag from LDS ring: ks {kh*16..+15}
    union { bf16x16 v; uint4 u[2]; } af;
    const bf16* ap = &Ab[kc & 3][(wv * 16 + mrow) * AST + kh * 16];
    af.u[0] = *(const uint4*)ap;
    af.u[1] = *(const uint4*)(ap + 8);

    // 4 B frags straight from transposed weights (L2/WGP$-resident)
    const int kb = kc << 5;
    union { bf16x16 v; uint4 u[2]; } b0, b1, b2, b3;
    {
      const bf16* w0 = wgtT + (size_t)(nbase +  0 + mrow) * K9 + kb + kh * 16;
      const bf16* w1 = wgtT + (size_t)(nbase + 16 + mrow) * K9 + kb + kh * 16;
      const bf16* w2 = wgtT + (size_t)(nbase + 32 + mrow) * K9 + kb + kh * 16;
      const bf16* w3 = wgtT + (size_t)(nbase + 48 + mrow) * K9 + kb + kh * 16;
      b0.u[0] = *(const uint4*)w0; b0.u[1] = *(const uint4*)(w0 + 8);
      b1.u[0] = *(const uint4*)w1; b1.u[1] = *(const uint4*)(w1 + 8);
      b2.u[0] = *(const uint4*)w2; b2.u[1] = *(const uint4*)(w2 + 8);
      b3.u[0] = *(const uint4*)w3; b3.u[1] = *(const uint4*)(w3 + 8);
    }

    acc0 = __builtin_amdgcn_wmma_f32_16x16x32_bf16(false, af.v, false, b0.v,
                                                   (short)0, acc0, false, false);
    acc1 = __builtin_amdgcn_wmma_f32_16x16x32_bf16(false, af.v, false, b1.v,
                                                   (short)0, acc1, false, false);
    acc2 = __builtin_amdgcn_wmma_f32_16x16x32_bf16(false, af.v, false, b2.v,
                                                   (short)0, acc2, false, false);
    acc3 = __builtin_amdgcn_wmma_f32_16x16x32_bf16(false, af.v, false, b3.v,
                                                   (short)0, acc3, false, false);
  }

  // ---- epilogue: C layout: VGPR r -> M = r + kh*8 ; N = lane&15 per j tile
  const int rowb = mbase + wv * 16;
#pragma unroll
  for (int j = 0; j < 4; ++j) {
    const int   nn   = nbase + j * 16 + mrow;
    const float badd = bias[nn];
    const f32x8 a = (j == 0) ? acc0 : (j == 1) ? acc1 : (j == 2) ? acc2 : acc3;
#pragma unroll
    for (int r = 0; r < 8; ++r) {
      const int g = rowb + r + kh * 8;
      out[(size_t)g * CH_ + nn] = (bf16)(a[r] + badd);
    }
  }
}

// ---------------------------------------------------------------------------
// GroupNorm(32 groups) + ReLU, in-place on bf16 [D,H,W,192]. One block per (d,g).
// ---------------------------------------------------------------------------
__global__ void lpn_gn_relu(bf16* __restrict__ x, const float* __restrict__ sc,
                            const float* __restrict__ bi, int H, int W)
{
  const int CG = CH_ / GROUPS_;                 // 6
  const int d = blockIdx.x / GROUPS_;
  const int g = blockIdx.x % GROUPS_;
  const int n = H * W * CG;
  const size_t base = (size_t)d * H * W * CH_;

  float s = 0.f, s2 = 0.f;
  for (int i = threadIdx.x; i < n; i += blockDim.x) {
    const int hw = i / CG;
    const int c  = g * CG + (i - hw * CG);
    const float v = (float)x[base + (size_t)hw * CH_ + c];
    s += v; s2 += v * v;
  }
  __shared__ float rs[256], rq[256];
  rs[threadIdx.x] = s; rq[threadIdx.x] = s2;
  __syncthreads();
  for (int t = 128; t > 0; t >>= 1) {
    if ((int)threadIdx.x < t) { rs[threadIdx.x] += rs[threadIdx.x + t];
                                rq[threadIdx.x] += rq[threadIdx.x + t]; }
    __syncthreads();
  }
  const float mu  = rs[0] / n;
  const float var = rq[0] / n - mu * mu;
  const float inv = rsqrtf(var + 1e-5f);

  for (int i = threadIdx.x; i < n; i += blockDim.x) {
    const int hw = i / CG;
    const int c  = g * CG + (i - hw * CG);
    const size_t idx = base + (size_t)hw * CH_ + c;
    const float v = ((float)x[idx] - mu) * inv * sc[c] + bi[c];
    x[idx] = (bf16)fmaxf(v, 0.f);
  }
}

// ---------------------------------------------------------------------------
// 1x1 heads + softmax + prediction generation (per voxel).
// Activation rows staged via async global->LDS copies (ASYNCcnt).
// ---------------------------------------------------------------------------
#define HB_ 128
__global__ __launch_bounds__(HB_)
void lpn_heads(const bf16* __restrict__ x,
               const float* __restrict__ cw, const float* __restrict__ cb,
               const float* __restrict__ rw, const float* __restrict__ rb,
               float* __restrict__ logits, float* __restrict__ regs,
               float* __restrict__ scores, float* __restrict__ locs,
               float* __restrict__ clsout,
               int D, int H, int W, float scale, int M)
{
  __shared__ __align__(16) bf16 xs[HB_ * CH_];

  const int t  = threadIdx.x;
  const int g0 = blockIdx.x * HB_;
  const int g  = g0 + t;
  const int gc = (g < M) ? g : (M - 1);          // clamp: keep async addr valid

  {
    const unsigned ldsoff = (unsigned)(size_t)(xs + t * CH_);
    const unsigned long long ga =
        (unsigned long long)(size_t)(x + (size_t)gc * CH_);
#pragma unroll
    for (int i = 0; i < CH_ * (int)sizeof(bf16); i += 16) {
      asm volatile("global_load_async_to_lds_b128 %0, %1, off"
                   :: "v"(ldsoff + i), "v"(ga + (unsigned long long)i)
                   : "memory");
    }
  }
  asm volatile("s_wait_asynccnt 0" ::: "memory");
  __syncthreads();

  if (g >= M) return;
  const bf16* xp = xs + t * CH_;
  float l0 = cb[0], l1 = cb[1];
  float r0 = rb[0], r1 = rb[1], r2 = rb[2];
  for (int c = 0; c < CH_; ++c) {
    const float v = (float)xp[c];
    l0 += v * cw[c * 2 + 0]; l1 += v * cw[c * 2 + 1];
    r0 += v * rw[c * 3 + 0]; r1 += v * rw[c * 3 + 1]; r2 += v * rw[c * 3 + 2];
  }
  logits[(size_t)g * 2 + 0] = l0; logits[(size_t)g * 2 + 1] = l1;
  regs[(size_t)g * 3 + 0] = r0; regs[(size_t)g * 3 + 1] = r1; regs[(size_t)g * 3 + 2] = r2;

  const float mx = fmaxf(l0, l1);
  const float e0 = __expf(l0 - mx), e1 = __expf(l1 - mx);
  const float p0 = e0 / (e0 + e1);

  const int xc = g % W, yc = (g / W) % H, dc = g / (W * H);
  const float lz = dc + 0.5f + r0, ly = yc + 0.5f + r1, lx = xc + 0.5f + r2;
  const bool valid = lz > 0.f && ly > 0.f && lx > 0.f &&
                     lz < (float)D && ly < (float)H && lx < (float)W;
  scores[g] = valid ? p0 : -1.0f;
  locs[(size_t)g * 3 + 0] = lz * ZSCALE_;
  locs[(size_t)g * 3 + 1] = ly * scale;
  locs[(size_t)g * 3 + 2] = lx * scale;
  clsout[g] = 0.0f;                              // N_CLS==1 -> argmax is always 0
}

// ---------------------------------------------------------------------------
// Per-voxel targets + focal/huber loss accumulation (atomics into accum[3]).
// ---------------------------------------------------------------------------
__global__ void lpn_losses(const float* __restrict__ logits, const float* __restrict__ regs,
                           const float* __restrict__ gtl, const int* __restrict__ gtc,
                           float* __restrict__ accum,
                           int D, int H, int W, float scale, int M)
{
  const int g = blockIdx.x * blockDim.x + threadIdx.x;
  if (g >= M) return;
  const int xc = g % W, yc = (g / W) % H, dc = g / (W * H);
  const float mz = (dc + 0.5f) * ZSCALE_;
  const float my = (yc + 0.5f) * scale;
  const float mx = (xc + 0.5f) * scale;

  float best = 1e30f; int bi = 0; float bz = 0.f, by = 0.f, bx = 0.f;
  for (int i = 0; i < 64; ++i) {
    const float fz = gtl[i * 3 + 0] * ZSCALE_;
    const float fy = gtl[i * 3 + 1];
    const float fx = gtl[i * 3 + 2];
    const float dz = fz - mz, dy = fy - my, dx = fx - mx;
    float d2 = dz * dz + dy * dy + dx * dx;
    if (!(fz >= 0.f && fy >= 0.f && fx >= 0.f)) d2 = 1e30f;
    if (d2 < best) { best = d2; bi = i; bz = dz; by = dy; bx = dx; }
  }
  const int clst = (best < 64.0f) ? gtc[bi] : 1;

  const float l0 = logits[(size_t)g * 2], l1 = logits[(size_t)g * 2 + 1];
  const float mm = fmaxf(l0, l1);
  const float e0 = __expf(l0 - mm), e1 = __expf(l1 - mm);
  const float p0 = e0 / (e0 + e1),  p1 = e1 / (e0 + e1);

  float cl = 0.f;
  {
    const float g0 = (clst == 0) ? 1.f : 0.f;
    float pt  = g0 * p0 + (1.f - g0) * (1.f - p0);
    float ptc = fminf(fmaxf(pt, 1e-9f), 1.f);
    cl += -(1.f - pt) * (1.f - pt) * logf(ptc);
    const float g1 = (clst == 1) ? 1.f : 0.f;
    pt  = g1 * p1 + (1.f - g1) * (1.f - p1);
    ptc = fminf(fmaxf(pt, 1e-9f), 1.f);
    cl += -(1.f - pt) * (1.f - pt) * logf(ptc);
  }
  atomicAdd(&accum[0], cl * 0.5f);

  if (clst != 1) {
    const float tg[3] = {bz / ZSCALE_, by / scale, bx / scale};
    const float pr[3] = {regs[(size_t)g * 3], regs[(size_t)g * 3 + 1], regs[(size_t)g * 3 + 2]};
    float h = 0.f;
    for (int j = 0; j < 3; ++j) {
      const float a = fabsf(pr[j] - tg[j]);
      const float q = fminf(a, 1.0f);
      h += 0.5f * q * q + (a - q);
    }
    atomicAdd(&accum[1], h / 3.f);
    atomicAdd(&accum[2], 1.f);
  }
}

// ---------------------------------------------------------------------------
// Pre-NMS candidate selection: 2048 bucket-maxima over all scores.
// ---------------------------------------------------------------------------
__global__ void lpn_cands(const float* __restrict__ sco, const float* __restrict__ loc,
                          const float* __restrict__ cls,
                          float* __restrict__ cs, float* __restrict__ cl,
                          float* __restrict__ cc, int total)
{
  const int b = blockIdx.x * blockDim.x + threadIdx.x;
  if (b >= 2048) return;
  const int stride = total / 2048;               // 42
  const int s0 = b * stride;
  float best = -1e30f; int bi = s0;
  for (int j = 0; j < stride; ++j) {
    const float v = sco[s0 + j];
    if (v > best) { best = v; bi = s0 + j; }
  }
  cs[b] = best;
  cl[b * 3 + 0] = loc[(size_t)bi * 3 + 0];
  cl[b * 3 + 1] = loc[(size_t)bi * 3 + 1];
  cl[b * 3 + 2] = loc[(size_t)bi * 3 + 2];
  cc[b] = cls[bi];
}

// ---------------------------------------------------------------------------
// Greedy NMS: single block, 512 sequential selections over 2048 candidates.
// Writes d_out: scores[0:512), locations[512:2048), classes[2048:2560)
// ---------------------------------------------------------------------------
__global__ void lpn_nms(const float* __restrict__ cs, const float* __restrict__ cl,
                        const float* __restrict__ cc, float* __restrict__ out)
{
  const int N = 2048;
  __shared__ float sv[512];
  __shared__ int   si[512];
  __shared__ unsigned char avail[2048];
  __shared__ float bl[3];
  __shared__ int   bidx;
  __shared__ float bval;
  __shared__ int   okf;
  __shared__ int   nsel;

  const int t = threadIdx.x;
  for (int i = t; i < N; i += 512) avail[i] = 1;
  if (t == 0) nsel = 0;
  __syncthreads();

  const float thr = 1.0f / (8.0f * 8.0f);

  for (int it = 0; it < 512; ++it) {
    float bv = -1e30f; int bi = 0;
    for (int i = t; i < N; i += 512) {
      const float v = avail[i] ? cs[i] : -1e30f;
      if (v > bv) { bv = v; bi = i; }
    }
    sv[t] = bv; si[t] = bi;
    __syncthreads();
    for (int s = 256; s > 0; s >>= 1) {
      if (t < s && sv[t + s] > sv[t]) { sv[t] = sv[t + s]; si[t] = si[t + s]; }
      __syncthreads();
    }
    if (t == 0) {
      bidx = si[0]; bval = sv[0];
      bl[0] = cl[si[0] * 3 + 0]; bl[1] = cl[si[0] * 3 + 1]; bl[2] = cl[si[0] * 3 + 2];
      okf = (sv[0] >= 0.2f) ? 1 : 0;
      avail[si[0]] = 0;
    }
    __syncthreads();
    for (int i = t; i < N; i += 512) {
      if (avail[i]) {
        const float dz = bl[0] - cl[i * 3 + 0];
        const float dy = bl[1] - cl[i * 3 + 1];
        const float dx = bl[2] - cl[i * 3 + 2];
        const float d2 = dz * dz + dy * dy + dx * dx;
        if (1.0f / fmaxf(d2, 1e-8f) >= thr) avail[i] = 0;
      }
    }
    if (t == 0) {
      const bool ok = okf != 0;
      out[it] = ok ? bval : -1.0f;
      float* L = out + 512 + (size_t)it * 3;
      L[0] = ok ? bl[0] / ZSCALE_ : -1.0f;
      L[1] = ok ? bl[1] : -1.0f;
      L[2] = ok ? bl[2] : -1.0f;
      if (ok) { out[2048 + nsel] = cc[bidx]; nsel++; }
    }
    __syncthreads();
  }
  if (t == 0) for (int i = nsel; i < 512; ++i) out[2048 + i] = -1.0f;
}

// ---------------------------------------------------------------------------
// Best match of predictions to GT. train = d_out + 2560 (64x3).
// ---------------------------------------------------------------------------
__global__ void lpn_bestmatch(const float* __restrict__ gtl, const float* __restrict__ out,
                              float* __restrict__ train)
{
  __shared__ int mid[512];
  const int t = threadIdx.x;
  const float* pl = out + 512;
  if (t < 512) {
    const float pz = pl[t * 3 + 0] * ZSCALE_;
    const float py = pl[t * 3 + 1];
    const float px = pl[t * 3 + 2];
    float best = 1e30f; int bi = 0;
    for (int i = 0; i < 64; ++i) {
      const float dz = pz - gtl[i * 3 + 0] * ZSCALE_;
      const float dy = py - gtl[i * 3 + 1];
      const float dx = px - gtl[i * 3 + 2];
      const float d2 = dz * dz + dy * dy + dx * dx;
      if (d2 < best) { best = d2; bi = i; }
    }
    mid[t] = (best <= 144.0f) ? bi : -1;
  }
  __syncthreads();
  if (t < 64) {
    int mi = 512;
    for (int p = 0; p < 512; ++p) if (mid[p] == t) { mi = p; break; }
    if (mi == 512) {
      train[t * 3 + 0] = gtl[t * 3 + 0];
      train[t * 3 + 1] = gtl[t * 3 + 1];
      train[t * 3 + 2] = gtl[t * 3 + 2];
    } else {
      train[t * 3 + 0] = pl[mi * 3 + 0];
      train[t * 3 + 1] = pl[mi * 3 + 1];
      train[t * 3 + 2] = pl[mi * 3 + 2];
    }
  }
}

__global__ void lpn_finalize(const float* __restrict__ accum, float* __restrict__ out,
                             float invM)
{
  if (threadIdx.x == 0) {
    out[2752] = accum[0] * invM;
    out[2753] = accum[1] / (accum[2] + 1e-8f);
  }
}

// ---------------------------------------------------------------------------
// Host orchestration
// ---------------------------------------------------------------------------
extern "C" void kernel_launch(void* const* d_in, const int* in_sizes, int n_in,
                              void* d_out, int out_size, void* d_ws, size_t ws_size,
                              hipStream_t stream)
{
  const float* feat[3] = {(const float*)d_in[0], (const float*)d_in[1], (const float*)d_in[2]};
  const float* gtl = (const float*)d_in[3];
  const int*   gtc = (const int*)d_in[4];
  // params["0"],["1"],["2"]: 20 tensors each, dict insertion order:
  // conv0_w,conv0_b,gn0_s,gn0_b, conv1_*, conv2_*, conv3_*, cls_w,cls_b,reg_w,reg_b
  auto P = [&](int l, int k) -> const float* { return (const float*)d_in[5 + l * 20 + k]; };

  const int D = 4, CIN = 256;
  const int Hs[3] = {128, 64, 32};
  const int offl[3] = {0, 65536, 81920};
  const int total = 86016;

  // ---- workspace bump allocator
  char* wsb = (char*)d_ws;
  size_t off = 0;
  auto alloc = [&](size_t bytes) -> void* {
    void* p = wsb + off;
    off = (off + bytes + 255) & ~(size_t)255;
    return p;
  };
  const int maxM = D * 128 * 128;                                  // 65536
  bf16* actA = (bf16*)alloc((size_t)maxM * 256 * sizeof(bf16));
  bf16* actB = (bf16*)alloc((size_t)maxM * CH_ * sizeof(bf16));
  bf16* wbf[3][4];
  for (int l = 0; l < 3; ++l)
    for (int i = 0; i < 4; ++i)
      wbf[l][i] = (bf16*)alloc((size_t)9 * (i == 0 ? 256 : 192) * CH_ * sizeof(bf16));
  float* logitsB = (float*)alloc((size_t)total * 2 * sizeof(float));
  float* regB    = (float*)alloc((size_t)total * 3 * sizeof(float));
  float* scoB    = (float*)alloc((size_t)total * sizeof(float));
  float* locB    = (float*)alloc((size_t)total * 3 * sizeof(float));
  float* clsB    = (float*)alloc((size_t)total * sizeof(float));
  float* candS   = (float*)alloc(2048 * sizeof(float));
  float* candL   = (float*)alloc(2048 * 3 * sizeof(float));
  float* candC   = (float*)alloc(2048 * sizeof(float));
  float* accum   = (float*)alloc(256);

  // ---- transpose-cast conv weights to bf16 [192][9*Cin] (every call)
  for (int l = 0; l < 3; ++l)
    for (int i = 0; i < 4; ++i) {
      const int K = 9 * (i == 0 ? 256 : 192);
      const int n = K * CH_;
      lpn_cast_transpose<<<(n + 255) / 256, 256, 0, stream>>>(P(l, i * 4), wbf[l][i], K, CH_);
    }
  lpn_zero_f32<<<1, 32, 0, stream>>>(accum, 3);

  float* out = (float*)d_out;

  for (int l = 0; l < 3; ++l) {
    const int H = Hs[l], W = Hs[l];
    const int M = D * H * W;                     // multiple of 128
    const float scale = (float)(4 << l);

    const int n0 = M * CIN;
    lpn_cast_bf16<<<(n0 + 255) / 256, 256, 0, stream>>>(feat[l], actA, n0);

    bf16* cur = actA;
    bf16* nxt = actB;
    for (int i = 0; i < 4; ++i) {
      dim3 grid(M / 128, CH_ / 64);
      if (i == 0)
        lpn_conv3x3_wmma<256><<<grid, 256, 0, stream>>>(cur, wbf[l][i], P(l, i * 4 + 1),
                                                        nxt, H, W, M);
      else
        lpn_conv3x3_wmma<192><<<grid, 256, 0, stream>>>(cur, wbf[l][i], P(l, i * 4 + 1),
                                                        nxt, H, W, M);
      lpn_gn_relu<<<D * GROUPS_, 256, 0, stream>>>(nxt, P(l, i * 4 + 2), P(l, i * 4 + 3), H, W);
      bf16* t = cur; cur = nxt; nxt = t;
    }

    lpn_heads<<<(M + HB_ - 1) / HB_, HB_, 0, stream>>>(
        cur, P(l, 16), P(l, 17), P(l, 18), P(l, 19),
        logitsB + (size_t)offl[l] * 2, regB + (size_t)offl[l] * 3,
        scoB + offl[l], locB + (size_t)offl[l] * 3, clsB + offl[l],
        D, H, W, scale, M);

    lpn_losses<<<(M + 255) / 256, 256, 0, stream>>>(
        logitsB + (size_t)offl[l] * 2, regB + (size_t)offl[l] * 3,
        gtl, gtc, accum, D, H, W, scale, M);
  }

  lpn_cands<<<8, 256, 0, stream>>>(scoB, locB, clsB, candS, candL, candC, total);
  lpn_nms<<<1, 512, 0, stream>>>(candS, candL, candC, out);
  lpn_bestmatch<<<1, 512, 0, stream>>>(gtl, out, out + 2560);
  lpn_finalize<<<1, 32, 0, stream>>>(accum, out, 1.0f / (float)total);
}
